// NetVLAD_39573828665809
// MI455X (gfx1250) — matrix operand order repaired and checked by reading, not compiled, not run on hardware
//
#include <hip/hip_runtime.h>
#include <math.h>

// ---------------- CDNA5 WMMA types ----------------
typedef __attribute__((ext_vector_type(16))) __bf16 v16bf;
typedef __attribute__((ext_vector_type(8)))  __bf16 v8bf;
typedef __attribute__((ext_vector_type(8)))  float  v8f;

#define BB 64       // batch
#define CC 128      // channels
#define KK 64       // clusters
#define NN 4096     // H*W
#define NT 128      // n-columns per pass
#define NTP 136     // padded row stride (n-major arrays)
#define CP  136     // padded row stride (c-major arrays)
#define NPASS 8     // passes per workgroup -> 1024 columns per WG
#define NCHUNK 4    // NN / (NT*NPASS)

union Frag  { v16bf v; v8bf h[2]; };
union Pack4 { __bf16 h[4]; uint2 u; };

// =====================================================================
// Kernel 1: fused  logits(WMMA) -> softmax -> VLAD aggregation (WMMA)
// grid = BB*NCHUNK workgroups of 256 threads (8 wave32).
// Each WG owns its [chunk][b] output slice -> plain stores, deterministic.
// =====================================================================
__global__ __launch_bounds__(256)
void netvlad_fused(const float* __restrict__ x,
                   const float* __restrict__ conv_w,
                   float* __restrict__ vlad_part,   // [NCHUNK][B][K][C]
                   float* __restrict__ asum_part)   // [NCHUNK][B][K]
{
    extern __shared__ char smem[];
    __bf16* w_lds    = (__bf16*)smem;            // [KK][CP]   conv_w, c-contig
    __bf16* xs       = w_lds + KK * CP;          // [CC][NTP]  x, n-contig (GEMM2 B)
    __bf16* xsT      = xs    + CC * NTP;         // [NT][CP]   x, c-contig (GEMM1 B)
    __bf16* a_lds    = xsT   + NT * CP;          // [KK][NTP]  softmax out (GEMM2 A)
    float*  asum_acc = (float*)(a_lds + KK * NTP);  // [KK]

    const int tid  = threadIdx.x;
    const int wv   = tid >> 5;      // wave 0..7
    const int lane = tid & 31;
    const int lm   = lane & 15;     // lane mod 16
    const int hi   = lane >> 4;     // lane half

    const int b     = blockIdx.x >> 2;
    const int chunk = blockIdx.x & 3;
    const int nbase = chunk * (NT * NPASS);

    // conv_w -> LDS (bf16), zero per-k mass accumulators (barrier below covers)
    for (int i = tid; i < KK * CC; i += 256) {
        const int k = i >> 7, c = i & 127;
        w_lds[k * CP + c] = (__bf16)conv_w[i];
    }
    if (tid < KK) asum_acc[tid] = 0.0f;

    v8f vacc[4] = {};   // persistent VLAD accumulators: 4 k-tiles x (16k x 16c)

    for (int p = 0; p < NPASS; ++p) {
        const int n0 = nbase + p * NT;

        // ---- stage x[b, :, n0:n0+NT] into LDS in both layouts (fp32->bf16)
        {
            const int c    = tid >> 1;
            const int half = tid & 1;
            const float* xrow = x + ((size_t)b * CC + c) * NN + n0;
            if (p + 1 < NPASS) __builtin_prefetch(xrow + NT, 0, 0);
            const float4* xr4 = (const float4*)xrow;
#pragma unroll
            for (int i = 0; i < 16; ++i) {
                const int j4 = half * 16 + i;
                const float4 v = xr4[j4];
                const int n = j4 * 4;
                Pack4 pk;
                pk.h[0] = (__bf16)v.x; pk.h[1] = (__bf16)v.y;
                pk.h[2] = (__bf16)v.z; pk.h[3] = (__bf16)v.w;
                *(uint2*)(&xs[c * NTP + n]) = pk.u;       // n-contiguous copy
                xsT[(n + 0) * CP + c] = pk.h[0];          // transposed copy
                xsT[(n + 1) * CP + c] = pk.h[1];
                xsT[(n + 2) * CP + c] = pk.h[2];
                xsT[(n + 3) * CP + c] = pk.h[3];
            }
        }
        __syncthreads();

        // ---- GEMM1: this wave computes all 64 logit rows for 16 n-columns
        v8f acc[4] = {};
#pragma unroll
        for (int cb = 0; cb < 4; ++cb) {             // K-loop over C in 32s
            Frag bfr;
            const __bf16* bp = &xsT[(wv * 16 + lm) * CP + cb * 32 + hi * 16];
            bfr.h[0] = *(const v8bf*)bp;
            bfr.h[1] = *(const v8bf*)(bp + 8);
#pragma unroll
            for (int kt = 0; kt < 4; ++kt) {
                Frag afr;
                const __bf16* ap = &w_lds[(kt * 16 + lm) * CP + cb * 32 + hi * 16];
                afr.h[0] = *(const v8bf*)ap;
                afr.h[1] = *(const v8bf*)(ap + 8);
                acc[kt] = __builtin_amdgcn_wmma_f32_16x16x32_bf16(
                    false, afr.v, false, bfr.v, (short)0, acc[kt], false, false);
            }
        }

        // ---- softmax over k=64 for this lane's column (cross-half via shfl)
        float m = -3.0e38f;
#pragma unroll
        for (int kt = 0; kt < 4; ++kt)
#pragma unroll
            for (int j = 0; j < 8; ++j) m = fmaxf(m, acc[kt][j]);
        m = fmaxf(m, __shfl_xor(m, 16, 32));
        float s = 0.0f;
#pragma unroll
        for (int kt = 0; kt < 4; ++kt)
#pragma unroll
            for (int j = 0; j < 8; ++j) {
                const float e = __expf(acc[kt][j] - m);
                acc[kt][j] = e;
                s += e;
            }
        s += __shfl_xor(s, 16, 32);
        const float inv = 1.0f / s;
        {
            const int ncol = wv * 16 + lm;
#pragma unroll
            for (int kt = 0; kt < 4; ++kt)
#pragma unroll
                for (int j = 0; j < 8; ++j) {
                    const int k = kt * 16 + hi * 8 + j;
                    a_lds[k * NTP + ncol] = (__bf16)(acc[kt][j] * inv);
                }
        }
        __syncthreads();

        // ---- per-k attention mass (one thread per k -> deterministic)
        if (tid < KK) {
            float ps = 0.0f;
            const __bf16* ar = &a_lds[tid * NTP];
            for (int j = 0; j < NT; ++j) ps += (float)ar[j];
            asum_acc[tid] += ps;
        }

        // ---- GEMM2: vacc[kt] += A(16k x 32n) * B(32n x 16c), c-tile = wave
#pragma unroll
        for (int nb = 0; nb < 4; ++nb) {             // K-loop over N in 32s
            Frag bfr;
            const __bf16* bp = &xs[(wv * 16 + lm) * NTP + nb * 32 + hi * 16];
            bfr.h[0] = *(const v8bf*)bp;
            bfr.h[1] = *(const v8bf*)(bp + 8);
#pragma unroll
            for (int kt = 0; kt < 4; ++kt) {
                Frag afr;
                const __bf16* ap = &a_lds[(kt * 16 + lm) * NTP + nb * 32 + hi * 16];
                afr.h[0] = *(const v8bf*)ap;
                afr.h[1] = *(const v8bf*)(ap + 8);
                vacc[kt] = __builtin_amdgcn_wmma_f32_16x16x32_bf16(
                    false, afr.v, false, bfr.v, (short)0, vacc[kt], false, false);
            }
        }
        __syncthreads();   // xs/xsT/a_lds reused next pass
    }

    // ---- write this WG's exclusive partial slice (no atomics)
    {
        const int c = wv * 16 + lm;
        float* dst = vlad_part + ((size_t)chunk * BB + b) * KK * CC;
#pragma unroll
        for (int kt = 0; kt < 4; ++kt)
#pragma unroll
            for (int j = 0; j < 8; ++j) {
                const int k = kt * 16 + hi * 8 + j;   // D layout: row = j + 8*hi
                dst[k * CC + c] = vacc[kt][j];
            }
    }
    if (tid < KK)
        asum_part[((size_t)chunk * BB + b) * KK + tid] = asum_acc[tid];
}

// =====================================================================
// Kernel 2: sum chunk partials, subtract asum*centroids, intra-L2 per k,
// global L2, write [B, K*C]. One block per batch.
// =====================================================================
__global__ __launch_bounds__(256)
void netvlad_finalize(const float* __restrict__ vlad_part,
                      const float* __restrict__ asum_part,
                      const float* __restrict__ centroids,
                      float* __restrict__ out)
{
    __shared__ float asum_s[KK];
    __shared__ __align__(16) float nv_lds[KK * CC];
    __shared__ float wsum[8];
    __shared__ float gscale_s;

    const int b    = blockIdx.x;
    const int tid  = threadIdx.x;
    const int wv   = tid >> 5;
    const int lane = tid & 31;

    if (tid < KK) {
        float s = 0.0f;
        for (int ch = 0; ch < NCHUNK; ++ch)
            s += asum_part[((size_t)ch * BB + b) * KK + tid];
        asum_s[tid] = s;
    }
    __syncthreads();

    float gpart = 0.0f;
#pragma unroll
    for (int kk = 0; kk < 8; ++kk) {
        const int k = wv * 8 + kk;
        const int c = lane * 4;
        float4 v = make_float4(0.f, 0.f, 0.f, 0.f);
        for (int ch = 0; ch < NCHUNK; ++ch) {
            const float4 pv = *(const float4*)(vlad_part +
                ((((size_t)ch * BB + b) * KK + k) * CC + c));
            v.x += pv.x; v.y += pv.y; v.z += pv.z; v.w += pv.w;
        }
        const float4 ce = *(const float4*)(centroids + k * CC + c);
        const float as = asum_s[k];
        float4 r;
        r.x = v.x - as * ce.x; r.y = v.y - as * ce.y;
        r.z = v.z - as * ce.z; r.w = v.w - as * ce.w;
        float ss = r.x * r.x + r.y * r.y + r.z * r.z + r.w * r.w;
#pragma unroll
        for (int off = 16; off >= 1; off >>= 1) ss += __shfl_xor(ss, off, 32);
        const float rinv = 1.0f / fmaxf(sqrtf(ss), 1e-12f);
        float4 nv;
        nv.x = r.x * rinv; nv.y = r.y * rinv;
        nv.z = r.z * rinv; nv.w = r.w * rinv;
        *(float4*)(&nv_lds[k * CC + c]) = nv;
        gpart += nv.x * nv.x + nv.y * nv.y + nv.z * nv.z + nv.w * nv.w;
    }
#pragma unroll
    for (int off = 16; off >= 1; off >>= 1) gpart += __shfl_xor(gpart, off, 32);
    if (lane == 0) wsum[wv] = gpart;
    __syncthreads();
    if (tid == 0) {
        float t = 0.0f;
        for (int i = 0; i < 8; ++i) t += wsum[i];
        gscale_s = 1.0f / fmaxf(sqrtf(t), 1e-12f);
    }
    __syncthreads();
    const float gs = gscale_s;
    const float4* src = (const float4*)nv_lds;
    float4* dst = (float4*)(out + (size_t)b * KK * CC);
    for (int i = tid; i < (KK * CC) / 4; i += 256) {
        float4 v = src[i];
        v.x *= gs; v.y *= gs; v.z *= gs; v.w *= gs;
        dst[i] = v;
    }
}

// =====================================================================
extern "C" void kernel_launch(void* const* d_in, const int* in_sizes, int n_in,
                              void* d_out, int out_size, void* d_ws, size_t ws_size,
                              hipStream_t stream) {
    (void)in_sizes; (void)n_in; (void)out_size; (void)ws_size;
    const float* x         = (const float*)d_in[0];  // [B,C,H,W] f32
    const float* conv_w    = (const float*)d_in[1];  // [K,C]     f32
    const float* centroids = (const float*)d_in[2];  // [K,C]     f32
    float* out = (float*)d_out;                      // [B, K*C]  f32

    float* vlad_part = (float*)d_ws;                             // 8 MB
    float* asum_part = vlad_part + (size_t)NCHUNK * BB * KK * CC; // 64 KB

    const size_t smem1 =
        (size_t)(KK * CP + CC * NTP + NT * CP + KK * NTP) * sizeof(__bf16)
        + KK * sizeof(float);   // ~102 KB -> needs dynamic-LDS opt-in

    hipFuncSetAttribute((const void*)netvlad_fused,
                        hipFuncAttributeMaxDynamicSharedMemorySize, (int)smem1);

    netvlad_fused<<<dim3(BB * NCHUNK), dim3(256), smem1, stream>>>(
        x, conv_w, vlad_part, asum_part);
    netvlad_finalize<<<dim3(BB), dim3(256), 0, stream>>>(
        vlad_part, asum_part, centroids, out);
}
